// CCA_14680198218180
// MI455X (gfx1250) — matrix operand order repaired and checked by reading, not compiled, not run on hardware
//
#include <hip/hip_runtime.h>
#include <hip/hip_bf16.h>

typedef __bf16 bf16_t;
typedef __attribute__((ext_vector_type(16))) __bf16 v16bf;
typedef __attribute__((ext_vector_type(8)))  float  v8f;

#define N_IMG 8
#define C_IN  512
#define C_QK  64
#define H_DIM 128
#define W_DIM 128
#define HWSZ  (H_DIM * W_DIM)
#define NEG_INF_F (-1000000000.0f)

__device__ __forceinline__ v8f wmma_bf16(v16bf a, v16bf b, v8f c) {
  // D = A(16x32 bf16) * B(32x16 bf16) + C(16x16 f32)
  return __builtin_amdgcn_wmma_f32_16x16x32_bf16(false, a, false, b, (short)0, c, false, false);
}

// A fragment (16x32, MxK) from LDS row-major [m][k]; pitch multiple of 8 halves
// -> two 16B-aligned ds_load_b128.
// ISA: lanes 0-15 row M=lane, K={0..7,16..23}; lanes 16-31 K={8..15,24..31}.
__device__ __forceinline__ v16bf load_a_frag(const bf16_t* A, int pitch, int m0, int k0, int lane) {
  const int r  = m0 + (lane & 15);
  const int kh = (lane >> 4) << 3;                  // 0 or 8
  const bf16_t* p = A + r * pitch + k0 + kh;
  union { uint4 u[2]; v16bf v; } cvt;
  cvt.u[0] = *(const uint4*)p;                      // K = kh .. kh+7
  cvt.u[1] = *(const uint4*)(p + 16);               // K = 16+kh .. 16+kh+7
  return cvt.v;
}

// B fragment (32x16, KxN) from LDS stored TRANSPOSED as [n][k] (pitch over k).
// Lane takes K={kb..kb+7, kb+16..kb+23} at column n (kb=0 or 8) = 2x ds_load_b128.
__device__ __forceinline__ v16bf load_b_frag_t(const bf16_t* Bt, int pitch, int k0, int n0, int lane) {
  const int n  = n0 + (lane & 15);
  const int kb = k0 + ((lane >> 4) << 3);           // 0 or 8
  const bf16_t* p = Bt + n * pitch + kb;
  union { uint4 u[2]; v16bf v; } cvt;
  cvt.u[0] = *(const uint4*)p;
  cvt.u[1] = *(const uint4*)(p + 16);
  return cvt.v;
}

union pack4 { uint2 u; bf16_t h[4]; };

// ---------------------------------------------------------------------------
// Stage 1: fused q/k/v projection. GEMM Y[o,p] = sum_c W[o,c] x[c,p] + b[o].
// Tile: M=64 out-channels, N=128 pixels, K=64 per LDS stage (8 WMMA/wave/stage).
// blockIdx = (pixel_tile, out_tile{0=q,1=k,2..9=v}, image)
// ---------------------------------------------------------------------------
__global__ __launch_bounds__(256) void proj_kernel(
    const float* __restrict__ x,
    const float* __restrict__ Wq, const float* __restrict__ bq,
    const float* __restrict__ Wk, const float* __restrict__ bk,
    const float* __restrict__ Wv, const float* __restrict__ bv,
    bf16_t* __restrict__ qo, bf16_t* __restrict__ ko, bf16_t* __restrict__ vo)
{
  __shared__ __align__(16) bf16_t lW[64][72];      // A: [m][k], 64x64 pad->72
  __shared__ __align__(16) bf16_t lXt[128][72];    // B transposed: [n][k]
  const int pt  = blockIdx.x;
  const int ot  = blockIdx.y;
  const int img = blockIdx.z;
  const int tid = threadIdx.x, lane = tid & 31, wave = tid >> 5;

  const float* Wmat; const float* bias; bf16_t* obuf; int orow0, och;
  if (ot == 0)      { Wmat = Wq; bias = bq; obuf = qo; orow0 = 0; och = C_QK; }
  else if (ot == 1) { Wmat = Wk; bias = bk; obuf = ko; orow0 = 0; och = C_QK; }
  else { int s = ot - 2; Wmat = Wv + (size_t)s * 64 * C_IN; bias = bv + s * 64;
         obuf = vo; orow0 = s * 64; och = C_IN; }

  const float* xin = x + (size_t)img * C_IN * HWSZ;
  const int p0  = pt * 128;
  const int wm  = wave & 3;            // M-tile 0..3
  const int wn0 = (wave >> 2) << 2;    // N-tile base 0 or 4

  v8f acc[4] = {};

  for (int k0 = 0; k0 < C_IN; k0 += 64) {
    // W tile 64x64: float4 global loads, packed b64 LDS stores (both contiguous)
    for (int i = tid; i < 64 * 16; i += 256) {
      int m = i >> 4, k4 = (i & 15) << 2;
      float4 f = *(const float4*)&Wmat[(size_t)m * C_IN + k0 + k4];
      pack4 pk;
      pk.h[0] = (bf16_t)f.x; pk.h[1] = (bf16_t)f.y;
      pk.h[2] = (bf16_t)f.z; pk.h[3] = (bf16_t)f.w;
      *(uint2*)&lW[m][k4] = pk.u;
    }
    // X tile 64x128 -> [n][k]: float4 global loads, 4 scalar transposed stores
    for (int i = tid; i < 64 * 32; i += 256) {
      int kk = i >> 5, n4 = (i & 31) << 2;
      float4 f = *(const float4*)&xin[(size_t)(k0 + kk) * HWSZ + p0 + n4];
      lXt[n4 + 0][kk] = (bf16_t)f.x;
      lXt[n4 + 1][kk] = (bf16_t)f.y;
      lXt[n4 + 2][kk] = (bf16_t)f.z;
      lXt[n4 + 3][kk] = (bf16_t)f.w;
    }
    __syncthreads();
#pragma unroll
    for (int kk = 0; kk < 64; kk += 32) {
      v16bf a = load_a_frag(&lW[0][0], 72, wm * 16, kk, lane);
#pragma unroll
      for (int t = 0; t < 4; ++t) {
        v16bf b = load_b_frag_t(&lXt[0][0], 72, kk, (wn0 + t) * 16, lane);
        acc[t] = wmma_bf16(a, b, acc[t]);
      }
    }
    __syncthreads();
  }

  const int crow = (lane >> 4) << 3, ccol = lane & 15;
#pragma unroll
  for (int t = 0; t < 4; ++t) {
    int n = p0 + (wn0 + t) * 16 + ccol;
#pragma unroll
    for (int r = 0; r < 8; ++r) {
      int m = wm * 16 + crow + r;
      obuf[((size_t)img * och + orow0 + m) * HWSZ + n] = (bf16_t)(acc[t][r] + bias[m]);
    }
  }
}

// ---------------------------------------------------------------------------
// Stage 2a: row logits. Per (n,h): S[w,v] = sum_c q[c,h,w]*k[c,h,v].
// M=w(128), N=v(128), K=c(64). Store bf16 erow[n][h][w][v].
// ---------------------------------------------------------------------------
__global__ __launch_bounds__(256) void row_logits_kernel(
    const bf16_t* __restrict__ q, const bf16_t* __restrict__ k, bf16_t* __restrict__ erow)
{
  __shared__ __align__(16) bf16_t lQ[128][72];     // A: [w][c]
  __shared__ __align__(16) bf16_t lKt[128][72];    // B transposed: [v][c]
  const int h = blockIdx.x, img = blockIdx.y;
  const int tid = threadIdx.x, lane = tid & 31, wave = tid >> 5;
  const size_t base = (size_t)img * C_QK * HWSZ + (size_t)h * W_DIM;

  // uint4 (8 bf16) global loads along spatial dim, transposed scalar LDS stores
  for (int i = tid; i < C_QK * 16; i += 256) {
    int c = i >> 4, w8 = (i & 15) << 3;
    union { uint4 u; bf16_t h[8]; } v8x;
    v8x.u = *(const uint4*)&q[base + (size_t)c * HWSZ + w8];
#pragma unroll
    for (int j = 0; j < 8; ++j) lQ[w8 + j][c] = v8x.h[j];
  }
  for (int i = tid; i < C_QK * 16; i += 256) {
    int c = i >> 4, v8i = (i & 15) << 3;
    union { uint4 u; bf16_t h[8]; } v8x;
    v8x.u = *(const uint4*)&k[base + (size_t)c * HWSZ + v8i];
#pragma unroll
    for (int j = 0; j < 8; ++j) lKt[v8i + j][c] = v8x.h[j];
  }
  __syncthreads();

  v8f acc[8] = {};
#pragma unroll
  for (int kk = 0; kk < C_QK; kk += 32) {
    v16bf a = load_a_frag(&lQ[0][0], 72, wave * 16, kk, lane);
#pragma unroll
    for (int t = 0; t < 8; ++t) {
      v16bf b = load_b_frag_t(&lKt[0][0], 72, kk, t * 16, lane);
      acc[t] = wmma_bf16(a, b, acc[t]);
    }
  }

  const int crow = (lane >> 4) << 3, ccol = lane & 15;
  const size_t obase = (((size_t)img * H_DIM + h) * W_DIM) * W_DIM;
#pragma unroll
  for (int t = 0; t < 8; ++t)
#pragma unroll
    for (int r = 0; r < 8; ++r) {
      int wq = wave * 16 + crow + r, v = t * 16 + ccol;
      erow[obase + (size_t)wq * W_DIM + v] = (bf16_t)acc[t][r];
    }
}

// ---------------------------------------------------------------------------
// Stage 2b: column logits. Per (n,w): S[h,g] = sum_c q[c,h,w]*k[c,g,w],
// diagonal h==g masked to NEG_INF. Store bf16 ecol[n][w][h][g].
// ---------------------------------------------------------------------------
__global__ __launch_bounds__(256) void col_logits_kernel(
    const bf16_t* __restrict__ q, const bf16_t* __restrict__ k, bf16_t* __restrict__ ecol)
{
  __shared__ __align__(16) bf16_t lQ[128][72];     // A: [h][c]
  __shared__ __align__(16) bf16_t lKt[128][72];    // B transposed: [g][c]
  const int wfix = blockIdx.x, img = blockIdx.y;
  const int tid = threadIdx.x, lane = tid & 31, wave = tid >> 5;
  const size_t base = (size_t)img * C_QK * HWSZ + wfix;

  for (int i = tid; i < C_QK * 128; i += 256) {
    int c = i >> 7, hq = i & 127;                  // strided over rows of the image
    lQ[hq][c] = q[base + (size_t)c * HWSZ + (size_t)hq * W_DIM];
  }
  for (int i = tid; i < C_QK * 128; i += 256) {
    int c = i >> 7, g = i & 127;
    lKt[g][c] = k[base + (size_t)c * HWSZ + (size_t)g * W_DIM];
  }
  __syncthreads();

  v8f acc[8] = {};
#pragma unroll
  for (int kk = 0; kk < C_QK; kk += 32) {
    v16bf a = load_a_frag(&lQ[0][0], 72, wave * 16, kk, lane);
#pragma unroll
    for (int t = 0; t < 8; ++t) {
      v16bf b = load_b_frag_t(&lKt[0][0], 72, kk, t * 16, lane);
      acc[t] = wmma_bf16(a, b, acc[t]);
    }
  }

  const int crow = (lane >> 4) << 3, ccol = lane & 15;
  const size_t obase = (((size_t)img * W_DIM + wfix) * H_DIM) * H_DIM;
#pragma unroll
  for (int t = 0; t < 8; ++t)
#pragma unroll
    for (int r = 0; r < 8; ++r) {
      int hq = wave * 16 + crow + r, g = t * 16 + ccol;
      float val = (hq == g) ? NEG_INF_F : acc[t][r];
      ecol[obase + (size_t)hq * H_DIM + g] = (bf16_t)val;
    }
}

// ---------------------------------------------------------------------------
// Stage 3: joint softmax over 256 logits (128 row + 128 col) per query pixel.
// One wave32 per query; b64-vectorized bf16 loads/stores; shfl_xor reductions.
// ---------------------------------------------------------------------------
__global__ __launch_bounds__(256) void softmax_kernel(
    bf16_t* __restrict__ erow, bf16_t* __restrict__ ecol)
{
  const int lane = threadIdx.x & 31;
  const int qid  = blockIdx.x * 8 + (threadIdx.x >> 5);   // (n*H + h)*W + w
  const int n = qid >> 14, h = (qid >> 7) & 127, w = qid & 127;
  const size_t rbase = (size_t)qid * W_DIM + lane * 4;
  const size_t cbase = (((size_t)n * W_DIM + w) * H_DIM + h) * H_DIM + lane * 4;

  pack4 pr, pc;
  pr.u = *(const uint2*)&erow[rbase];
  pc.u = *(const uint2*)&ecol[cbase];

  float v[8];
#pragma unroll
  for (int i = 0; i < 4; ++i) { v[i] = (float)pr.h[i]; v[4 + i] = (float)pc.h[i]; }

  float m = v[0];
#pragma unroll
  for (int i = 1; i < 8; ++i) m = fmaxf(m, v[i]);
  for (int off = 16; off > 0; off >>= 1) m = fmaxf(m, __shfl_xor(m, off, 32));

  float s = 0.f;
#pragma unroll
  for (int i = 0; i < 8; ++i) { v[i] = __expf(v[i] - m); s += v[i]; }
  for (int off = 16; off > 0; off >>= 1) s += __shfl_xor(s, off, 32);
  const float inv = 1.0f / s;

#pragma unroll
  for (int i = 0; i < 4; ++i) { pr.h[i] = (bf16_t)(v[i] * inv); pc.h[i] = (bf16_t)(v[4 + i] * inv); }
  *(uint2*)&erow[rbase] = pr.u;
  *(uint2*)&ecol[cbase] = pc.u;
}

// ---------------------------------------------------------------------------
// Stage 4a: row output. Per (n,h): O[c,w] = sum_v V[c,h,v] * A_row[w,v].
// M=c (chunks of 128), N=w(128), K=v(128). f32 partial into otmp[n][c][h][w].
// ---------------------------------------------------------------------------
__global__ __launch_bounds__(256) void out_row_kernel(
    const bf16_t* __restrict__ arow, const bf16_t* __restrict__ vbuf,
    float* __restrict__ otmp)
{
  __shared__ __align__(16) bf16_t lV[128][136];    // A: [c][v]
  __shared__ __align__(16) bf16_t lAt[128][136];   // B transposed: [w][v]
  const int h = blockIdx.x, img = blockIdx.y;
  const int tid = threadIdx.x, lane = tid & 31, wave = tid >> 5;

  const size_t abase = (((size_t)img * H_DIM + h) * W_DIM) * W_DIM;
  for (int i = tid; i < 128 * 16; i += 256) {      // 16x uint4 per row
    int wq = i >> 4, v8i = (i & 15) << 3;
    *(uint4*)&lAt[wq][v8i] = *(const uint4*)&arow[abase + (size_t)wq * W_DIM + v8i];
  }

  const size_t vbase = (size_t)img * C_IN * HWSZ + (size_t)h * W_DIM;
  const int crow = (lane >> 4) << 3, ccol = lane & 15;

  for (int c0 = 0; c0 < C_IN; c0 += 128) {
    for (int i = tid; i < 128 * 16; i += 256) {
      int cc = i >> 4, v8i = (i & 15) << 3;
      *(uint4*)&lV[cc][v8i] = *(const uint4*)&vbuf[vbase + (size_t)(c0 + cc) * HWSZ + v8i];
    }
    __syncthreads();

    v8f acc[8] = {};
#pragma unroll
    for (int kk = 0; kk < 128; kk += 32) {
      v16bf a = load_a_frag(&lV[0][0], 136, wave * 16, kk, lane);
#pragma unroll
      for (int t = 0; t < 8; ++t) {
        v16bf b = load_b_frag_t(&lAt[0][0], 136, kk, t * 16, lane);
        acc[t] = wmma_bf16(a, b, acc[t]);
      }
    }

#pragma unroll
    for (int t = 0; t < 8; ++t)
#pragma unroll
      for (int r = 0; r < 8; ++r) {
        int m = wave * 16 + crow + r;   // channel within chunk
        int n = t * 16 + ccol;          // w
        otmp[(((size_t)img * C_IN + c0 + m) * H_DIM + h) * W_DIM + n] = acc[t][r];
      }
    __syncthreads();
  }
}

// ---------------------------------------------------------------------------
// Stage 4b: column output + epilogue. Per (n,w): O[c,h] = sum_g V[c,g,w]*A_col[h,g];
// final out = gamma*(row+col) + x.
// ---------------------------------------------------------------------------
__global__ __launch_bounds__(256) void out_col_kernel(
    const bf16_t* __restrict__ acol, const bf16_t* __restrict__ vbuf,
    const float* __restrict__ otmp, const float* __restrict__ x,
    const float* __restrict__ gamma, float* __restrict__ out)
{
  __shared__ __align__(16) bf16_t lV[128][136];    // A: [c][g]
  __shared__ __align__(16) bf16_t lAt[128][136];   // B transposed: [h][g]
  const int wfix = blockIdx.x, img = blockIdx.y;
  const int tid = threadIdx.x, lane = tid & 31, wave = tid >> 5;
  const float g0 = gamma[0];

  const size_t abase = (((size_t)img * W_DIM + wfix) * H_DIM) * H_DIM;
  for (int i = tid; i < 128 * 16; i += 256) {
    int hq = i >> 4, g8 = (i & 15) << 3;
    *(uint4*)&lAt[hq][g8] = *(const uint4*)&acol[abase + (size_t)hq * H_DIM + g8];
  }

  const size_t vbase = (size_t)img * C_IN * HWSZ + wfix;
  const int crow = (lane >> 4) << 3, ccol = lane & 15;

  for (int c0 = 0; c0 < C_IN; c0 += 128) {
    for (int i = tid; i < 128 * 128; i += 256) {
      int cc = i >> 7, g = i & 127;                // strided gather over g
      lV[cc][g] = vbuf[vbase + (size_t)(c0 + cc) * HWSZ + (size_t)g * W_DIM];
    }
    __syncthreads();

    v8f acc[8] = {};
#pragma unroll
    for (int kk = 0; kk < 128; kk += 32) {
      v16bf a = load_a_frag(&lV[0][0], 136, wave * 16, kk, lane);
#pragma unroll
      for (int t = 0; t < 8; ++t) {
        v16bf b = load_b_frag_t(&lAt[0][0], 136, kk, t * 16, lane);
        acc[t] = wmma_bf16(a, b, acc[t]);
      }
    }

#pragma unroll
    for (int t = 0; t < 8; ++t)
#pragma unroll
      for (int r = 0; r < 8; ++r) {
        int m = wave * 16 + crow + r;   // channel within chunk
        int n = t * 16 + ccol;          // h
        size_t idx = (((size_t)img * C_IN + c0 + m) * H_DIM + n) * W_DIM + wfix;
        out[idx] = g0 * (acc[t][r] + otmp[idx]) + x[idx];
      }
    __syncthreads();
  }
}

// ---------------------------------------------------------------------------
extern "C" void kernel_launch(void* const* d_in, const int* in_sizes, int n_in,
                              void* d_out, int out_size, void* d_ws, size_t ws_size,
                              hipStream_t stream) {
  (void)in_sizes; (void)n_in; (void)out_size; (void)ws_size;
  const float* x     = (const float*)d_in[0];
  const float* Wq    = (const float*)d_in[1];
  const float* bq    = (const float*)d_in[2];
  const float* Wk    = (const float*)d_in[3];
  const float* bk    = (const float*)d_in[4];
  const float* Wv    = (const float*)d_in[5];
  const float* bv    = (const float*)d_in[6];
  const float* gamma = (const float*)d_in[7];
  float* out = (float*)d_out;

  char* ws = (char*)d_ws;
  size_t off = 0;
  bf16_t* qw = (bf16_t*)(ws + off); off += (size_t)N_IMG * C_QK * HWSZ * 2;   // 16.8 MB
  bf16_t* kw = (bf16_t*)(ws + off); off += (size_t)N_IMG * C_QK * HWSZ * 2;   // 16.8 MB
  bf16_t* vw = (bf16_t*)(ws + off); off += (size_t)N_IMG * C_IN * HWSZ * 2;   // 134 MB
  bf16_t* er = (bf16_t*)(ws + off); off += (size_t)N_IMG * HWSZ * W_DIM * 2;  // 33.5 MB
  bf16_t* ec = (bf16_t*)(ws + off); off += (size_t)N_IMG * HWSZ * H_DIM * 2;  // 33.5 MB
  float*  tp = (float*)(ws + off);                                            // 268 MB

  proj_kernel<<<dim3(HWSZ / 128, 10, N_IMG), 256, 0, stream>>>(
      x, Wq, bq, Wk, bk, Wv, bv, qw, kw, vw);
  row_logits_kernel<<<dim3(H_DIM, N_IMG), 256, 0, stream>>>(qw, kw, er);
  col_logits_kernel<<<dim3(W_DIM, N_IMG), 256, 0, stream>>>(qw, kw, ec);
  softmax_kernel<<<dim3((N_IMG * HWSZ) / 8), 256, 0, stream>>>(er, ec);
  out_row_kernel<<<dim3(H_DIM, N_IMG), 256, 0, stream>>>(er, vw, tp);
  out_col_kernel<<<dim3(W_DIM, N_IMG), 256, 0, stream>>>(ec, vw, tp, x, gamma, out);
}
